// InverseFoldingEncoder_317827580826
// MI455X (gfx1250) — matrix operand order repaired and checked by reading, not compiled, not run on hardware
//
#include <hip/hip_runtime.h>
#include <hip/hip_bf16.h>

typedef _Float16 f16;
typedef __attribute__((ext_vector_type(16))) _Float16 v16h;
typedef __attribute__((ext_vector_type(8)))  float    v8f;
typedef __attribute__((ext_vector_type(4)))  unsigned int v4u;
typedef __attribute__((ext_vector_type(8)))  int      v8i;
typedef __attribute__((ext_vector_type(4)))  int      v4i;

#define NN   10000
#define EE   300000
#define TOPKE 30
#define BN_RSQ 0.9999950000374996f   // 1/sqrt(1+1e-5)

// tanh-approx GELU: 0.5x(1+tanh(.79788(x+.044715x^3))) == x*sigmoid(2y)
// -> single v_exp_f32 (TRANS co-executes with WMMA), branch-free.
__device__ __forceinline__ float gelu_f(float x) {
  float u = 1.5957691216057308f * (x + 0.044715f * x * x * x);
  return x / (1.0f + __expf(-u));
}

// ---------------------------------------------------------------------------
// TDM: async 2D f16 tile load Global->LDS (Tensor Data Mover, CDNA5 ISA ch.8).
// D# group0: count=1 | lds_addr | global_addr | type=2
// D# group1: data_size=2B, pad_enable + (pad_interval,pad_amount) so TDM
//            deposits rows at a padded LDS stride (bank-conflict-free frags),
//            tensor_dim0/1, tile_dim0/1, tensor_dim0_stride.
// Issued by one wave; completion via s_wait_tensorcnt 0.
// This toolchain exposes the 6-arg builtin (g0, g1, g2, g3, g4, cpol).
// ---------------------------------------------------------------------------
__device__ __forceinline__ void tdm_load_2d_f16(
    unsigned lds_byte_addr, const f16* gsrc,
    unsigned rowlen,        // elements (f16) per row == tile_dim0 == tensor_dim0
    unsigned rows_valid,    // tensor_dim1 (OOB rows read as zero)
    unsigned rows_tile,     // tile_dim1
    unsigned stride_elts,   // tensor_dim0_stride (f16 elements)
    unsigned pad_interval,  // pad after 2^(v+1) DWORDs
    unsigned pad_amount)    // pad (v+1) DWORDs
{
  unsigned long long ga = (unsigned long long)(size_t)gsrc;
  v4u g0;
  g0.x = 1u;                                           // count=1, user D#
  g0.y = lds_byte_addr;                                // bits 63:32
  g0.z = (unsigned)(ga & 0xffffffffull);               // global_addr lo
  g0.w = (unsigned)((ga >> 32) & 0x01ffffffull) | (2u << 30);  // type=2
  v8i g1;
  g1[0] = (int)((1u << 16)                             // data_size = 2 bytes
              | (1u << 20)                             // pad_enable
              | (pad_interval << 22) | (pad_amount << 25));
  g1[1] = (int)((rowlen & 0xffffu) << 16);             // tensor_dim0[15:0]
  g1[2] = (int)(((rowlen >> 16) & 0xffffu) | ((rows_valid & 0xffffu) << 16));
  g1[3] = (int)(((rows_valid >> 16) & 0xffffu) | ((rowlen & 0xffffu) << 16)); // tile_dim0
  g1[4] = (int)(rows_tile & 0xffffu);                  // tile_dim1 (tile_dim2=0)
  g1[5] = (int)stride_elts;                            // dim0_stride[31:0]
  g1[6] = 0;                                           // dim0_stride hi / dim1_stride lo
  g1[7] = 0;
  v4i gz4 = {0, 0, 0, 0};
  v8i gz8 = {0, 0, 0, 0, 0, 0, 0, 0};
  __builtin_amdgcn_tensor_load_to_lds(g0, g1, gz4, gz4, gz8, 0);
}

// ---------------------------------------------------------------------------
// Wave-cooperative 64x128 GEMM slice with v_wmma_f32_16x16x32_f16.
// Block = 256 threads = 8 waves; wave w owns rows [16*(w>>1),+16) and cols
// [(w&1)*64,+64) as 4 tiles. A in LDS (f16, 64 x lda). Weights WT in global
// f16, stored transposed [128][ldw]; this call covers K-range [k0, k0+32*kt).
// Fragment k order: contiguous, k = 16*(lane/16) + s for both A and B (wave32).
// ---------------------------------------------------------------------------
__device__ __forceinline__ void wmma_gemm64x128(
    const f16* __restrict__ As, int lda,
    const f16* __restrict__ WT, int ldw, int k0, int ktiles,
    v8f acc[4])
{
  const int lane   = threadIdx.x & 31;
  const int wave   = threadIdx.x >> 5;
  const int mtile  = wave >> 1;
  const int nbase0 = (wave & 1) * 64;
  const int row    = mtile * 16 + (lane & 15);
  const int asel   = (lane >> 4) << 4;   // 0 or 16
  const int bsel   = (lane >> 4) << 4;   // 0 or 16
  const f16* Arow  = As + row * lda;
  for (int kt = 0; kt < ktiles; ++kt) {
    const int kb = kt * 32;
    v16h a;
#pragma unroll
    for (int i = 0; i < 8; ++i) a[i]     = Arow[kb + asel + i];
#pragma unroll
    for (int i = 0; i < 8; ++i) a[8 + i] = Arow[kb + asel + 8 + i];
#pragma unroll
    for (int t = 0; t < 4; ++t) {
      const int n = nbase0 + t * 16 + (lane & 15);
      const f16* Wp = WT + (size_t)n * ldw + k0 + kb + bsel;
      v16h b;
#pragma unroll
      for (int i = 0; i < 16; ++i) b[i] = Wp[i];
      acc[t] = __builtin_amdgcn_wmma_f32_16x16x32_f16(
          false, a, false, b, (short)0, acc[t], false, false);
    }
  }
}

#define EPI_SETUP()                                   \
  const int lane   = threadIdx.x & 31;                \
  const int wave   = threadIdx.x >> 5;                \
  const int mtile  = wave >> 1;                       \
  const int nbase0 = (wave & 1) * 64;                 \
  const int msub   = ((lane >> 4) << 3)

// ---------------------------------------------------------------------------
// Kernel 1: edge FFN.  z += BN(W2 @ gelu(W1 @ [s[src]|s[dst]|z] + b1) + b2)
// z-block staged by TDM (async) while threads gather s[src]/s[dst].
// ---------------------------------------------------------------------------
__global__ __launch_bounds__(256) void k_edge_ffn(
    const f16* __restrict__ s16, f16* __restrict__ z16, float* __restrict__ z_cur,
    const int* __restrict__ src, const int* __restrict__ dst,
    const f16* __restrict__ W1T, const float* __restrict__ b1,
    const f16* __restrict__ W2T, const float* __restrict__ b2,
    const float* __restrict__ gam, const float* __restrict__ bet)
{
  __shared__ f16 As[64 * 264];   // K-cols 0..255 : [s[src] | s[dst]]
  __shared__ f16 Zs[64 * 136];   // K-cols 256..383 : z rows (TDM, padded stride)
  __shared__ f16 Hs[64 * 136];
  const int tid   = threadIdx.x;
  const int ebase = blockIdx.x * 64;
  const int rows  = (EE - ebase < 64) ? (EE - ebase) : 64;
  __builtin_prefetch(W1T + (tid << 6), 0, 1);
  __builtin_prefetch(W2T + (tid << 6), 0, 1);

  if (tid < 32)   // one wave issues the tensor DMA: 64x128 f16, LDS stride 136
    tdm_load_2d_f16((unsigned)(size_t)Zs, z16 + (size_t)ebase * 128,
                    128, (unsigned)rows, 64, 128, /*pad*/5, 3);

  for (int i = tid; i < 64 * 256; i += 256) {
    int r = i >> 8, c = i & 255;
    int e = ebase + r; if (e >= EE) e = EE - 1;
    f16 v = (c < 128) ? s16[(size_t)src[e] * 128 + c]
                      : s16[(size_t)dst[e] * 128 + (c - 128)];
    As[r * 264 + c] = v;
  }
  if (tid < 32) __builtin_amdgcn_s_wait_tensorcnt(0);
  __syncthreads();

  v8f acc[4] = {};
  wmma_gemm64x128(As, 264, W1T, 384, 0,   8, acc);
  wmma_gemm64x128(Zs, 136, W1T, 384, 256, 4, acc);

  EPI_SETUP();
#pragma unroll
  for (int t = 0; t < 4; ++t) {
    int n = nbase0 + t * 16 + (lane & 15);
    float bn = b1[n];
#pragma unroll
    for (int r = 0; r < 8; ++r)
      Hs[(mtile * 16 + msub + r) * 136 + n] = (f16)gelu_f(acc[t][r] + bn);
  }
  __syncthreads();

  v8f acc2[4] = {};
  wmma_gemm64x128(Hs, 136, W2T, 128, 0, 4, acc2);

#pragma unroll
  for (int t = 0; t < 4; ++t) {
    int n = nbase0 + t * 16 + (lane & 15);
    float bn = b2[n], g = gam[n] * BN_RSQ, bb = bet[n];
#pragma unroll
    for (int r = 0; r < 8; ++r) {
      int e = ebase + mtile * 16 + msub + r;
      if (e < EE) {
        size_t o = (size_t)e * 128 + n;
        float y = (acc2[t][r] + bn) * g + bb + z_cur[o];
        z_cur[o] = y;
        z16[o]   = (f16)y;
      }
    }
  }
}

// ---------------------------------------------------------------------------
// Kernel 2: attention weight MLP: [s[dst]|s[src]|z] -> H -> H -> NH logits
// ---------------------------------------------------------------------------
__global__ __launch_bounds__(256) void k_attn_w(
    const f16* __restrict__ s16, const f16* __restrict__ z16,
    const int* __restrict__ src, const int* __restrict__ dst,
    const f16* __restrict__ W1T, const float* __restrict__ b1,
    const f16* __restrict__ W2T, const float* __restrict__ b2,
    const f16* __restrict__ W3T, const float* __restrict__ b3,
    float* __restrict__ aw)
{
  __shared__ f16 As[64 * 264];
  __shared__ f16 Zs[64 * 136];
  __shared__ f16 Hs[64 * 136];
  __shared__ f16 H2[64 * 136];
  const int tid   = threadIdx.x;
  const int ebase = blockIdx.x * 64;
  const int rows  = (EE - ebase < 64) ? (EE - ebase) : 64;
  __builtin_prefetch(W1T + (tid << 6), 0, 1);

  if (tid < 32)
    tdm_load_2d_f16((unsigned)(size_t)Zs, z16 + (size_t)ebase * 128,
                    128, (unsigned)rows, 64, 128, 5, 3);

  for (int i = tid; i < 64 * 256; i += 256) {
    int r = i >> 8, c = i & 255;
    int e = ebase + r; if (e >= EE) e = EE - 1;
    f16 v = (c < 128) ? s16[(size_t)dst[e] * 128 + c]
                      : s16[(size_t)src[e] * 128 + (c - 128)];
    As[r * 264 + c] = v;
  }
  if (tid < 32) __builtin_amdgcn_s_wait_tensorcnt(0);
  __syncthreads();

  v8f acc[4] = {};
  wmma_gemm64x128(As, 264, W1T, 384, 0,   8, acc);
  wmma_gemm64x128(Zs, 136, W1T, 384, 256, 4, acc);

  EPI_SETUP();
#pragma unroll
  for (int t = 0; t < 4; ++t) {
    int n = nbase0 + t * 16 + (lane & 15);
    float bn = b1[n];
#pragma unroll
    for (int r = 0; r < 8; ++r)
      Hs[(mtile * 16 + msub + r) * 136 + n] = (f16)gelu_f(acc[t][r] + bn);
  }
  __syncthreads();

  v8f acc2[4] = {};
  wmma_gemm64x128(Hs, 136, W2T, 128, 0, 4, acc2);
#pragma unroll
  for (int t = 0; t < 4; ++t) {
    int n = nbase0 + t * 16 + (lane & 15);
    float bn = b2[n];
#pragma unroll
    for (int r = 0; r < 8; ++r)
      H2[(mtile * 16 + msub + r) * 136 + n] = (f16)gelu_f(acc2[t][r] + bn);
  }
  __syncthreads();

  // Final Linear(128 -> 4): one (edge, head) dot per thread
  {
    int el = tid >> 2, h = tid & 3;
    float d = b3[h];
    const f16* wrow = W3T + h * 128;
    const f16* hrow = H2 + el * 136;
#pragma unroll 8
    for (int k = 0; k < 128; ++k) d += (float)hrow[k] * (float)wrow[k];
    int e = ebase + el;
    if (e < EE) aw[(size_t)e * 4 + h] = d;
  }
}

// ---------------------------------------------------------------------------
// Kernel 3: attention value MLP: [s[src]|z] -> H -> H -> D
// ---------------------------------------------------------------------------
__global__ __launch_bounds__(256) void k_attn_v(
    const f16* __restrict__ s16, const f16* __restrict__ z16,
    const int* __restrict__ src,
    const f16* __restrict__ W1T, const float* __restrict__ b1,
    const f16* __restrict__ W2T, const float* __restrict__ b2,
    const f16* __restrict__ W3T, const float* __restrict__ b3,
    f16* __restrict__ av16)
{
  __shared__ f16 As[64 * 136];   // K-cols 0..127 : s[src]
  __shared__ f16 Zs[64 * 136];   // K-cols 128..255 : z (TDM)
  __shared__ f16 Hs[64 * 136];
  __shared__ f16 H2[64 * 136];
  const int tid   = threadIdx.x;
  const int ebase = blockIdx.x * 64;
  const int rows  = (EE - ebase < 64) ? (EE - ebase) : 64;
  __builtin_prefetch(W1T + (tid << 6), 0, 1);

  if (tid < 32)
    tdm_load_2d_f16((unsigned)(size_t)Zs, z16 + (size_t)ebase * 128,
                    128, (unsigned)rows, 64, 128, 5, 3);

  for (int i = tid; i < 64 * 128; i += 256) {
    int r = i >> 7, c = i & 127;
    int e = ebase + r; if (e >= EE) e = EE - 1;
    As[r * 136 + c] = s16[(size_t)src[e] * 128 + c];
  }
  if (tid < 32) __builtin_amdgcn_s_wait_tensorcnt(0);
  __syncthreads();

  v8f acc[4] = {};
  wmma_gemm64x128(As, 136, W1T, 256, 0,   4, acc);
  wmma_gemm64x128(Zs, 136, W1T, 256, 128, 4, acc);

  EPI_SETUP();
#pragma unroll
  for (int t = 0; t < 4; ++t) {
    int n = nbase0 + t * 16 + (lane & 15);
    float bn = b1[n];
#pragma unroll
    for (int r = 0; r < 8; ++r)
      Hs[(mtile * 16 + msub + r) * 136 + n] = (f16)gelu_f(acc[t][r] + bn);
  }
  __syncthreads();

  v8f acc2[4] = {};
  wmma_gemm64x128(Hs, 136, W2T, 128, 0, 4, acc2);
#pragma unroll
  for (int t = 0; t < 4; ++t) {
    int n = nbase0 + t * 16 + (lane & 15);
    float bn = b2[n];
#pragma unroll
    for (int r = 0; r < 8; ++r)
      H2[(mtile * 16 + msub + r) * 136 + n] = (f16)gelu_f(acc2[t][r] + bn);
  }
  __syncthreads();

  v8f acc3[4] = {};
  wmma_gemm64x128(H2, 136, W3T, 128, 0, 4, acc3);
#pragma unroll
  for (int t = 0; t < 4; ++t) {
    int n = nbase0 + t * 16 + (lane & 15);
    float bn = b3[n];
#pragma unroll
    for (int r = 0; r < 8; ++r) {
      int e = ebase + mtile * 16 + msub + r;
      if (e < EE) av16[(size_t)e * 128 + n] = (f16)(acc3[t][r] + bn);
    }
  }
}

// ---------------------------------------------------------------------------
// Kernel 4: per-dst segment softmax (30 edges, 4 heads) + weighted aggregate
// ---------------------------------------------------------------------------
__global__ __launch_bounds__(128) void k_softmax_agg(
    const float* __restrict__ aw, const f16* __restrict__ av16,
    f16* __restrict__ agg16)
{
  __shared__ float P[128];
  const int n = blockIdx.x, tid = threadIdx.x;
  const int base = n * TOPKE;
  if (tid < TOPKE * 4) P[tid] = aw[(size_t)base * 4 + tid];
  __syncthreads();
  if (tid < 4) {
    float mx = -3.4e38f;
    for (int e = 0; e < TOPKE; ++e) mx = fmaxf(mx, P[e * 4 + tid]);
    float s = 0.f;
    for (int e = 0; e < TOPKE; ++e) { float v = __expf(P[e * 4 + tid] - mx); P[e * 4 + tid] = v; s += v; }
    float inv = 1.f / s;
    for (int e = 0; e < TOPKE; ++e) P[e * 4 + tid] *= inv;
  }
  __syncthreads();
  const int d = tid;
  float a0 = 0.f, a1 = 0.f, a2 = 0.f, a3 = 0.f;
  for (int e = 0; e < TOPKE; ++e) {
    float v = (float)av16[(size_t)(base + e) * 128 + d];
    a0 += P[e * 4 + 0] * v; a1 += P[e * 4 + 1] * v;
    a2 += P[e * 4 + 2] * v; a3 += P[e * 4 + 3] * v;
  }
  size_t o = (size_t)n * 512 + d;
  agg16[o]       = (f16)a0; agg16[o + 128] = (f16)a1;
  agg16[o + 256] = (f16)a2; agg16[o + 384] = (f16)a3;
}

// ---------------------------------------------------------------------------
// Kernel 5: node update.  s += BN(ao @ agg); s += BN(f2 @ gelu(f1 @ s))
// agg tile (64x512 f16, contiguous) staged entirely by TDM.
// ---------------------------------------------------------------------------
__global__ __launch_bounds__(256) void k_node_update(
    const f16* __restrict__ agg16, float* __restrict__ s_cur,
    const f16* __restrict__ WoT, const float* __restrict__ bo,
    const float* __restrict__ gam_o, const float* __restrict__ bet_o,
    const f16* __restrict__ W1T, const float* __restrict__ b1,
    const f16* __restrict__ W2T, const float* __restrict__ b2,
    const float* __restrict__ gam_f, const float* __restrict__ bet_f)
{
  __shared__ f16 As[64 * 520];
  __shared__ f16 Smid[64 * 136];
  const int tid  = threadIdx.x;
  const int nb   = blockIdx.x * 64;
  const int rows = (NN - nb < 64) ? (NN - nb) : 64;
  __builtin_prefetch(WoT + (tid << 6), 0, 1);

  if (tid < 32) {  // 64x512 f16 tile, LDS row stride 520 via TDM padding
    tdm_load_2d_f16((unsigned)(size_t)As, agg16 + (size_t)nb * 512,
                    512, (unsigned)rows, 64, 512, /*pad*/7, 3);
    __builtin_amdgcn_s_wait_tensorcnt(0);
  }
  __syncthreads();

  v8f acc[4] = {};
  wmma_gemm64x128(As, 520, WoT, 512, 0, 16, acc);

  EPI_SETUP();
#pragma unroll
  for (int t = 0; t < 4; ++t) {
    int n = nbase0 + t * 16 + (lane & 15);
    float bn = bo[n], g = gam_o[n] * BN_RSQ, bb = bet_o[n];
#pragma unroll
    for (int r = 0; r < 8; ++r) {
      int m = mtile * 16 + msub + r;
      int node = nb + m; if (node >= NN) node = NN - 1;
      float smid = s_cur[(size_t)node * 128 + n] + (acc[t][r] + bn) * g + bb;
      Smid[m * 136 + n] = (f16)smid;
    }
  }
  __syncthreads();  // all waves done with As; reuse as Hs

  f16* Hs = As;
  v8f acc2[4] = {};
  wmma_gemm64x128(Smid, 136, W1T, 128, 0, 4, acc2);
#pragma unroll
  for (int t = 0; t < 4; ++t) {
    int n = nbase0 + t * 16 + (lane & 15);
    float bn = b1[n];
#pragma unroll
    for (int r = 0; r < 8; ++r)
      Hs[(mtile * 16 + msub + r) * 136 + n] = (f16)gelu_f(acc2[t][r] + bn);
  }
  __syncthreads();

  v8f acc3[4] = {};
  wmma_gemm64x128(Hs, 136, W2T, 128, 0, 4, acc3);
#pragma unroll
  for (int t = 0; t < 4; ++t) {
    int n = nbase0 + t * 16 + (lane & 15);
    float bn = b2[n], g = gam_f[n] * BN_RSQ, bb = bet_f[n];
#pragma unroll
    for (int r = 0; r < 8; ++r) {
      int m = mtile * 16 + msub + r;
      int node = nb + m;
      if (node < NN) {
        float y = (acc3[t][r] + bn) * g + bb + (float)Smid[m * 136 + n];
        s_cur[(size_t)node * 128 + n] = y;
      }
    }
  }
}

// ---------------------------------------------------------------------------
// Utility kernels
// ---------------------------------------------------------------------------
__global__ void k_copy_f32(const float* __restrict__ x, float* __restrict__ y, long n) {
  long i = (long)blockIdx.x * blockDim.x + threadIdx.x;
  if (i < n) y[i] = x[i];
}
__global__ void k_cvt16(const float* __restrict__ x, f16* __restrict__ y, long n) {
  long i = (long)blockIdx.x * blockDim.x + threadIdx.x;
  if (i < n) y[i] = (f16)x[i];
}
// w: [K][Nn] f32 row-major  ->  wt: [Nn][K] f16 (transposed, contiguous B frags)
__global__ void k_cvt_wT(const float* __restrict__ w, f16* __restrict__ wt, int K, int Nn) {
  int i = blockIdx.x * 256 + threadIdx.x;
  if (i < K * Nn) {
    int k = i / Nn, n = i - k * Nn;
    wt[(size_t)n * K + k] = (f16)w[i];
  }
}

// ---------------------------------------------------------------------------
extern "C" void kernel_launch(void* const* d_in, const int* in_sizes, int n_in,
                              void* d_out, int out_size, void* d_ws, size_t ws_size,
                              hipStream_t stream)
{
  (void)in_sizes; (void)n_in; (void)out_size; (void)ws_size;
  const float* s_in = (const float*)d_in[0];
  const float* z_in = (const float*)d_in[1];
  const int*   eidx = (const int*)d_in[2];
  const int* src = eidx;
  const int* dst = eidx + EE;

  char* wsp = (char*)d_ws;
  auto take = [&](size_t bytes) -> char* {
    char* p = wsp; wsp += (bytes + 255) & ~(size_t)255; return p;
  };
  float* z_cur = (float*)take((size_t)EE * 128 * 4);
  float* s_cur = (float*)take((size_t)NN * 128 * 4);
  f16*   z16   = (f16*)  take((size_t)EE * 128 * 2);
  f16*   s16   = (f16*)  take((size_t)NN * 128 * 2);
  f16*   av16  = (f16*)  take((size_t)EE * 128 * 2);
  float* aw    = (float*)take((size_t)EE * 4   * 4);
  f16*   agg16 = (f16*)  take((size_t)NN * 512 * 2);

  // Weight matrices per layer (d_in order): ew1 ew2 aw1 aw2 aw3 av1 av2 av3 ao f1 f2
  static const int KD[11] = {384,128,384,128,128,256,128,128,512,128,128};
  static const int ND[11] = {128,128,128,128,  4,128,128,128,128,128,128};
  static const int WI[11] = {0,2,4,6,8,10,12,14,16,18,20};
  f16* wT[3][11];
  for (int l = 0; l < 3; ++l)
    for (int j = 0; j < 11; ++j) {
      int K = KD[j], Nn = ND[j], tot = K * Nn;
      wT[l][j] = (f16*)take((size_t)tot * 2);
      k_cvt_wT<<<(tot + 255) / 256, 256, 0, stream>>>(
          (const float*)d_in[3 + 28 * l + WI[j]], wT[l][j], K, Nn);
    }

  const long n_s = (long)NN * 128, n_z = (long)EE * 128;
  k_copy_f32<<<(int)((n_s + 255) / 256), 256, 0, stream>>>(s_in, s_cur, n_s);
  k_copy_f32<<<(int)((n_z + 255) / 256), 256, 0, stream>>>(z_in, z_cur, n_z);
  k_cvt16   <<<(int)((n_s + 255) / 256), 256, 0, stream>>>(s_in, s16, n_s);
  k_cvt16   <<<(int)((n_z + 255) / 256), 256, 0, stream>>>(z_in, z16, n_z);

  const int EB = (EE + 63) / 64;
  const int NB = (NN + 63) / 64;

  for (int l = 0; l < 3; ++l) {
    auto P = [&](int off) -> const float* { return (const float*)d_in[3 + 28 * l + off]; };
    if (l > 0)
      k_cvt16<<<(int)((n_s + 255) / 256), 256, 0, stream>>>(s_cur, s16, n_s);

    k_edge_ffn<<<EB, 256, 0, stream>>>(s16, z16, z_cur, src, dst,
        wT[l][0], P(1), wT[l][1], P(3), P(22), P(23));
    k_attn_w<<<EB, 256, 0, stream>>>(s16, z16, src, dst,
        wT[l][2], P(5), wT[l][3], P(7), wT[l][4], P(9), aw);
    k_attn_v<<<EB, 256, 0, stream>>>(s16, z16, src,
        wT[l][5], P(11), wT[l][6], P(13), wT[l][7], P(15), av16);
    k_softmax_agg<<<NN, 128, 0, stream>>>(aw, av16, agg16);
    k_node_update<<<NB, 256, 0, stream>>>(agg16, s_cur,
        wT[l][8], P(17), P(24), P(25),
        wT[l][9], P(19), wT[l][10], P(21), P(26), P(27));
  }

  float* out = (float*)d_out;
  k_copy_f32<<<(int)((n_s + 255) / 256), 256, 0, stream>>>(s_cur, out, n_s);
  k_copy_f32<<<(int)((n_z + 255) / 256), 256, 0, stream>>>(z_cur, out + n_s, n_z);
}
